// ResBlock_A_90142773609280
// MI455X (gfx1250) — matrix operand order repaired and checked by reading, not compiled, not run on hardware
//
#include <hip/hip_runtime.h>

typedef __attribute__((ext_vector_type(16))) _Float16 v16h;
typedef __attribute__((ext_vector_type(8)))  _Float16 v8h;
typedef __attribute__((ext_vector_type(8)))  float    v8f;

#define NPTS   65536
#define KNBR   32
#define CIN    128
#define HDIM   64
#define ODIM   256
#define NKPT   15
#define KPEXT_INV 20.0f   // 1/0.05
#define BN_EPS 1e-5f
#define SLOPE  0.1f

// ---------------- workspace layout (bytes) ----------------
#define WS_FEATSH  ((size_t)0)                    // [N,128] f16  = 16 MB
#define WS_H1H     ((size_t)16u << 20)            // [N,64]  f16  =  8 MB (pre-BN conv1)
#define WS_Y1H     ((size_t)24u << 20)            // [N,64]  f16  =  8 MB
#define WS_H2H     ((size_t)32u << 20)            // [N,256] f16  = 32 MB
#define WS_W1T     (((size_t)65u << 20) + 0x00000) // [64][128] f16
#define WS_W2T     (((size_t)65u << 20) + 0x08000) // [256][64] f16
#define WS_WSCT    (((size_t)65u << 20) + 0x10000) // [256][128] f16
#define WS_WKT     (((size_t)65u << 20) + 0x20000) // [64][960] f16
#define WS_SC1     (((size_t)65u << 20) + 0x40000) // [64] f32
#define WS_SH1     (((size_t)65u << 20) + 0x41000)
#define WS_SC2     (((size_t)65u << 20) + 0x42000)
#define WS_SH2     (((size_t)65u << 20) + 0x43000)
#define WS_P1      (((size_t)65u << 20) + 0x50000) // [1024][128] f32 = 512 KB
#define WS_P2      (((size_t)65u << 20) + 0xD0000) // [1024][512] f32 = 2 MB

__device__ __forceinline__ float lrelu(float v) { return v >= 0.f ? v : SLOPE * v; }

// ---------------- WMMA helpers (CDNA5 wave32 layouts) ----------------
__device__ __forceinline__ v8f wmma16(v16h a, v16h b, v8f c) {
  return __builtin_amdgcn_wmma_f32_16x16x32_f16(false, a, false, b, (short)0, c,
                                                false, false);
}

// A fragment: src row-major [M][lda] f16. lane<16: M=lane, K = k0..k0+15
//             lane>=16: M=lane-16, K = k0+16..k0+31
__device__ __forceinline__ v16h load_a_frag(const _Float16* src, int lda, int m0, int k0) {
  int lane = threadIdx.x & 31;
  int m  = m0 + (lane & 15);
  int kb = k0 + (lane & 16);                 // +0 or +16
  v8h lo = *(const v8h*)(src + (size_t)m * lda + kb);
  v8h hi = *(const v8h*)(src + (size_t)m * lda + kb + 8);
  return __builtin_shufflevector(lo, hi, 0,1,2,3,4,5,6,7,8,9,10,11,12,13,14,15);
}

// B fragment from pre-transposed weights Wt row-major [Nout][ldk] f16.
// lane<16: N=n0+lane, K = k0..k0+15 ; lane>=16: N=n0+lane-16, K = k0+16..k0+31
__device__ __forceinline__ v16h load_b_fragT(const _Float16* Wt, int ldk, int n0, int k0) {
  int lane = threadIdx.x & 31;
  int n  = n0 + (lane & 15);
  int kb = k0 + (lane & 16);                 // +0 or +16
  v8h lo = *(const v8h*)(Wt + (size_t)n * ldk + kb);
  v8h hi = *(const v8h*)(Wt + (size_t)n * ldk + kb + 8);
  return __builtin_shufflevector(lo, hi, 0,1,2,3,4,5,6,7,8,9,10,11,12,13,14,15);
}

// B fragment from row-major tile X[K][ldn] of raw (pre-BN) f16, applying
// per-column BN scale/shift + lrelu on the fly.
__device__ __forceinline__ v16h load_b_frag_bn(const _Float16* X, int ldn, int n0, int k0,
                                               const float* __restrict__ scale,
                                               const float* __restrict__ shift) {
  int lane = threadIdx.x & 31;
  int n  = n0 + (lane & 15);
  int kb = k0 + (lane & 16);
  float sc = scale[n], sh = shift[n];
  v16h b;
#pragma unroll
  for (int j = 0; j < 16; ++j) {
    float v = (float)X[(size_t)(kb + j) * ldn + n] * sc + sh;
    b[j] = (_Float16)lrelu(v);
  }
  return b;
}

// ---------------- prep kernels ----------------
__global__ void k_cvt_f16(const float* __restrict__ src, _Float16* __restrict__ dst, int n) {
  for (int i = blockIdx.x * blockDim.x + threadIdx.x; i < n; i += gridDim.x * blockDim.x)
    dst[i] = (_Float16)src[i];
}

// dst[c][r] = src[r][c] ; src is [R][Cc]
__global__ void k_transpose_f16(const float* __restrict__ src, _Float16* __restrict__ dst,
                                int R, int Cc) {
  int total = R * Cc;
  for (int i = blockIdx.x * blockDim.x + threadIdx.x; i < total; i += gridDim.x * blockDim.x) {
    int r = i / Cc, c = i % Cc;
    dst[(size_t)c * R + r] = (_Float16)src[i];
  }
}

// WkT[d][p*64+h] = Wk[p][h][d]
__global__ void k_wkt(const float* __restrict__ wk, _Float16* __restrict__ dst) {
  int total = HDIM * (NKPT * HDIM);
  for (int i = blockIdx.x * blockDim.x + threadIdx.x; i < total; i += gridDim.x * blockDim.x) {
    int d = i / (NKPT * HDIM), j = i % (NKPT * HDIM);
    dst[i] = (_Float16)wk[(size_t)j * HDIM + d];
  }
}

// ---------------- GEMM1: featsH[N,128] @ W1t -> h1h[N,64] (+b1) + BN partials ----------------
__global__ void __launch_bounds__(128)
k_gemm1(const _Float16* __restrict__ A, const _Float16* __restrict__ Wt,
        const float* __restrict__ bias, _Float16* __restrict__ out,
        float* __restrict__ part) {
  __shared__ float red[8][64];
  __shared__ float redq[8][64];
  int w = threadIdx.x >> 5, lane = threadIdx.x & 31;
  int m0 = blockIdx.x * 64 + w * 16;
  v8f acc[4];
  v8f z = {0.f,0.f,0.f,0.f,0.f,0.f,0.f,0.f};
#pragma unroll
  for (int t = 0; t < 4; ++t) acc[t] = z;
#pragma unroll
  for (int ks = 0; ks < 4; ++ks) {
    v16h a = load_a_frag(A, CIN, m0, ks * 32);
#pragma unroll
    for (int t = 0; t < 4; ++t)
      acc[t] = wmma16(a, load_b_fragT(Wt, CIN, t * 16, ks * 32), acc[t]);
  }
  int rowoff = (lane & 16) >> 1, cl = lane & 15;
  int slot = w * 2 + ((lane & 16) >> 4);
#pragma unroll
  for (int t = 0; t < 4; ++t) {
    int col = t * 16 + cl;
    float b = bias[col], s = 0.f, q = 0.f;
#pragma unroll
    for (int r = 0; r < 8; ++r) {
      float v = acc[t][r] + b;
      int row = m0 + r + rowoff;
      out[(size_t)row * HDIM + col] = (_Float16)v;
      s += v; q += v * v;
    }
    red[slot][col] = s; redq[slot][col] = q;
  }
  __syncthreads();
  if (threadIdx.x < 64) {
    int col = threadIdx.x;
    float s = 0.f, q = 0.f;
#pragma unroll
    for (int i = 0; i < 8; ++i) { s += red[i][col]; q += redq[i][col]; }
    part[(size_t)blockIdx.x * 128 + col]      = s;
    part[(size_t)blockIdx.x * 128 + 64 + col] = q;
  }
}

// ---------------- BN stats reduce: partials -> scale/shift ----------------
__global__ void k_stats(const float* __restrict__ part, int nblk, int cols,
                        const float* __restrict__ g, const float* __restrict__ be,
                        float* __restrict__ scale, float* __restrict__ shift) {
  int col = threadIdx.x;
  if (col >= cols) return;
  float s = 0.f, q = 0.f;
  for (int b = 0; b < nblk; ++b) {
    s += part[(size_t)b * 2 * cols + col];
    q += part[(size_t)b * 2 * cols + cols + col];
  }
  float mu  = s / (float)NPTS;
  float var = q / (float)NPTS - mu * mu;
  float rs  = rsqrtf(var + BN_EPS);
  float sc  = g[col] * rs;
  scale[col] = sc;
  shift[col] = be[col] - mu * sc;
}

// ---------------- KPConv: async gather + influence WMMA + [16,960]@[960,64] WMMA --------
__global__ void __launch_bounds__(128)
k_kpconv(const float* __restrict__ points, const int* __restrict__ nbrs,
         const float* __restrict__ kp, const _Float16* __restrict__ h1,
         const float* __restrict__ scale1, const float* __restrict__ shift1,
         const _Float16* __restrict__ WkT, _Float16* __restrict__ y1h) {
  __shared__ _Float16 inflA[4][16 * 32];   // per-wave A tile [p][k]
  __shared__ _Float16 xnB[4][32 * 64];     // per-wave gathered raw h1 rows [k][h]
  __shared__ _Float16 WF[16][NKPT * HDIM]; // [point][p*64+h]
  __shared__ float kps[48];

  int tid = threadIdx.x, w = tid >> 5, lane = tid & 31;
  if (tid < NKPT * 3) kps[tid] = kp[tid];
  __syncthreads();

  int blk = blockIdx.x;
  int rowoff = (lane & 16) >> 1, cl = lane & 15;

  for (int it = 0; it < 4; ++it) {
    int pt = w * 4 + it;
    int n  = blk * 16 + pt;
    int idx = nbrs[(size_t)n * KNBR + lane];

    // scattered row gather straight into LDS via async DMA (ASYNCcnt path);
    // instruction offset applies to both LDS and memory address per ISA.
    {
      unsigned ldsb = (unsigned)(unsigned long long)(const void*)&xnB[w][lane * HDIM];
      unsigned long long ga = (unsigned long long)(const void*)(h1 + (size_t)idx * HDIM);
      asm volatile(
        "global_load_async_to_lds_b128 %0, %1, off\n\t"
        "global_load_async_to_lds_b128 %0, %1, off offset:16\n\t"
        "global_load_async_to_lds_b128 %0, %1, off offset:32\n\t"
        "global_load_async_to_lds_b128 %0, %1, off offset:48"
        :: "v"(ldsb), "v"(ga) : "memory");
    }

    float pcx = points[n * 3 + 0], pcy = points[n * 3 + 1], pcz = points[n * 3 + 2];
    float rx = points[(size_t)idx * 3 + 0] - pcx;
    float ry = points[(size_t)idx * 3 + 1] - pcy;
    float rz = points[(size_t)idx * 3 + 2] - pcz;
#pragma unroll
    for (int p = 0; p < NKPT; ++p) {
      float dx = rx - kps[p * 3 + 0];
      float dy = ry - kps[p * 3 + 1];
      float dz = rz - kps[p * 3 + 2];
      float d = sqrtf(dx * dx + dy * dy + dz * dz);
      float f = 1.f - d * KPEXT_INV;
      inflA[w][p * 32 + lane] = (_Float16)(f > 0.f ? f : 0.f);
    }
    inflA[w][15 * 32 + lane] = (_Float16)0.f;   // pad row

    v16h a = load_a_frag(&inflA[w][0], 32, 0, 0);
    asm volatile("s_wait_asynccnt 0x0" ::: "memory");   // gather complete in LDS
#pragma unroll
    for (int t = 0; t < 4; ++t) {
      v8f c = {0.f,0.f,0.f,0.f,0.f,0.f,0.f,0.f};
      c = wmma16(a, load_b_frag_bn(&xnB[w][0], HDIM, t * 16, 0, scale1, shift1), c);
      int col = t * 16 + cl;
#pragma unroll
      for (int r = 0; r < 8; ++r) {
        int p = r + rowoff;
        if (p < NKPT) WF[pt][p * HDIM + col] = (_Float16)c[r];
      }
    }
  }
  __syncthreads();

  // [16 points, 960] @ WkT(960,64): each wave owns one 16-col tile
  v8f acc = {0.f,0.f,0.f,0.f,0.f,0.f,0.f,0.f};
  for (int ks = 0; ks < 30; ++ks) {
    v16h a = load_a_frag(&WF[0][0], NKPT * HDIM, 0, ks * 32);
    v16h b = load_b_fragT(WkT, NKPT * HDIM, w * 16, ks * 32);
    acc = wmma16(a, b, acc);
  }
  int col = w * 16 + cl;
#pragma unroll
  for (int r = 0; r < 8; ++r) {
    int prow = blk * 16 + r + rowoff;
    y1h[(size_t)prow * HDIM + col] = (_Float16)lrelu(acc[r]);
  }
}

// ---------------- GEMM3: y1h[N,64] @ W2t -> h2h[N,256] (+b2) + BN partials ----------------
__global__ void __launch_bounds__(128)
k_gemm3(const _Float16* __restrict__ A, const _Float16* __restrict__ Wt,
        const float* __restrict__ bias, _Float16* __restrict__ out,
        float* __restrict__ part) {
  __shared__ float red[8][256];
  __shared__ float redq[8][256];
  int w = threadIdx.x >> 5, lane = threadIdx.x & 31;
  int m0 = blockIdx.x * 64 + w * 16;
  v8f acc[16];
  v8f z = {0.f,0.f,0.f,0.f,0.f,0.f,0.f,0.f};
#pragma unroll
  for (int t = 0; t < 16; ++t) acc[t] = z;
#pragma unroll
  for (int ks = 0; ks < 2; ++ks) {
    v16h a = load_a_frag(A, HDIM, m0, ks * 32);
#pragma unroll
    for (int t = 0; t < 16; ++t)
      acc[t] = wmma16(a, load_b_fragT(Wt, HDIM, t * 16, ks * 32), acc[t]);
  }
  int rowoff = (lane & 16) >> 1, cl = lane & 15;
  int slot = w * 2 + ((lane & 16) >> 4);
#pragma unroll
  for (int t = 0; t < 16; ++t) {
    int col = t * 16 + cl;
    float b = bias[col], s = 0.f, q = 0.f;
#pragma unroll
    for (int r = 0; r < 8; ++r) {
      float v = acc[t][r] + b;
      int row = m0 + r + rowoff;
      out[(size_t)row * ODIM + col] = (_Float16)v;
      s += v; q += v * v;
    }
    red[slot][col] = s; redq[slot][col] = q;
  }
  __syncthreads();
  for (int c0 = threadIdx.x; c0 < 256; c0 += 128) {
    float s = 0.f, q = 0.f;
#pragma unroll
    for (int i = 0; i < 8; ++i) { s += red[i][c0]; q += redq[i][c0]; }
    part[(size_t)blockIdx.x * 512 + c0]       = s;
    part[(size_t)blockIdx.x * 512 + 256 + c0] = q;
  }
}

// ---------------- final: BN2-apply + lrelu + featsH@WscT shortcut + lrelu ----------------
__global__ void __launch_bounds__(128)
k_final(const _Float16* __restrict__ A, const _Float16* __restrict__ Wt,
        const _Float16* __restrict__ h2, const float* __restrict__ scale,
        const float* __restrict__ shift, float* __restrict__ out) {
  int w = threadIdx.x >> 5, lane = threadIdx.x & 31;
  int m0 = blockIdx.x * 64 + w * 16;
  v8f acc[16];
  v8f z = {0.f,0.f,0.f,0.f,0.f,0.f,0.f,0.f};
#pragma unroll
  for (int t = 0; t < 16; ++t) acc[t] = z;
#pragma unroll
  for (int ks = 0; ks < 4; ++ks) {
    v16h a = load_a_frag(A, CIN, m0, ks * 32);
#pragma unroll
    for (int t = 0; t < 16; ++t)
      acc[t] = wmma16(a, load_b_fragT(Wt, CIN, t * 16, ks * 32), acc[t]);
  }
  int rowoff = (lane & 16) >> 1, cl = lane & 15;
#pragma unroll
  for (int t = 0; t < 16; ++t) {
    int col = t * 16 + cl;
    float sc = scale[col], sh = shift[col];
#pragma unroll
    for (int r = 0; r < 8; ++r) {
      int row = m0 + r + rowoff;
      size_t o = (size_t)row * ODIM + col;
      float y2 = lrelu((float)h2[o] * sc + sh);
      out[o] = lrelu(y2 + acc[t][r]);
    }
  }
}

// ---------------- host ----------------
extern "C" void kernel_launch(void* const* d_in, const int* in_sizes, int n_in,
                              void* d_out, int out_size, void* d_ws, size_t ws_size,
                              hipStream_t stream) {
  (void)in_sizes; (void)n_in; (void)out_size; (void)ws_size;
  const float* feats = (const float*)d_in[0];
  const float* points = (const float*)d_in[1];
  const float* W1  = (const float*)d_in[2];
  const float* b1  = (const float*)d_in[3];
  const float* g1  = (const float*)d_in[4];
  const float* be1 = (const float*)d_in[5];
  const float* kp  = (const float*)d_in[6];
  const float* Wk  = (const float*)d_in[7];
  const float* W2  = (const float*)d_in[8];
  const float* b2  = (const float*)d_in[9];
  const float* g2  = (const float*)d_in[10];
  const float* be2 = (const float*)d_in[11];
  const float* Wsc = (const float*)d_in[12];
  const int* nbrs  = (const int*)d_in[13];
  float* out = (float*)d_out;

  char* ws = (char*)d_ws;
  _Float16* featsH = (_Float16*)(ws + WS_FEATSH);
  _Float16* h1h    = (_Float16*)(ws + WS_H1H);
  _Float16* y1h    = (_Float16*)(ws + WS_Y1H);
  _Float16* h2h    = (_Float16*)(ws + WS_H2H);
  _Float16* W1t    = (_Float16*)(ws + WS_W1T);
  _Float16* W2t    = (_Float16*)(ws + WS_W2T);
  _Float16* WscT   = (_Float16*)(ws + WS_WSCT);
  _Float16* WkT    = (_Float16*)(ws + WS_WKT);
  float* sc1 = (float*)(ws + WS_SC1);
  float* sh1 = (float*)(ws + WS_SH1);
  float* sc2 = (float*)(ws + WS_SC2);
  float* sh2 = (float*)(ws + WS_SH2);
  float* p1  = (float*)(ws + WS_P1);
  float* p2  = (float*)(ws + WS_P2);

  // prep: f16 conversions + weight transposes
  hipLaunchKernelGGL(k_cvt_f16, dim3(4096), dim3(256), 0, stream, feats, featsH, NPTS * CIN);
  hipLaunchKernelGGL(k_transpose_f16, dim3(32),  dim3(256), 0, stream, W1,  W1t,  CIN,  HDIM);
  hipLaunchKernelGGL(k_transpose_f16, dim3(64),  dim3(256), 0, stream, W2,  W2t,  HDIM, ODIM);
  hipLaunchKernelGGL(k_transpose_f16, dim3(128), dim3(256), 0, stream, Wsc, WscT, CIN,  ODIM);
  hipLaunchKernelGGL(k_wkt, dim3(240), dim3(256), 0, stream, Wk, WkT);

  // stage 1: conv1 (pre-BN) + BN stats
  hipLaunchKernelGGL(k_gemm1, dim3(NPTS / 64), dim3(128), 0, stream, featsH, W1t, b1, h1h, p1);
  hipLaunchKernelGGL(k_stats, dim3(1), dim3(64), 0, stream, p1, NPTS / 64, HDIM, g1, be1, sc1, sh1);

  // stage 2: KPConv (BN1-apply + lrelu fused into the gather consumer) + lrelu
  hipLaunchKernelGGL(k_kpconv, dim3(NPTS / 16), dim3(128), 0, stream,
                     points, nbrs, kp, h1h, sc1, sh1, WkT, y1h);

  // stage 3: conv2 + BN stats
  hipLaunchKernelGGL(k_gemm3, dim3(NPTS / 64), dim3(128), 0, stream, y1h, W2t, b2, h2h, p2);
  hipLaunchKernelGGL(k_stats, dim3(1), dim3(256), 0, stream, p2, NPTS / 64, ODIM, g2, be2, sc2, sh2);

  // stage 4: BN apply + shortcut GEMM + residual + lrelu
  hipLaunchKernelGGL(k_final, dim3(NPTS / 64), dim3(128), 0, stream,
                     featsH, WscT, h2h, sc2, sh2, out);
}
